// InformationGainKeyframeSelector_49392123904656
// MI455X (gfx1250) — matrix operand order repaired and checked by reading, not compiled, not run on hardware
//
#include <hip/hip_runtime.h>
#include <cstdint>
#include <cmath>

#define B_    256
#define T_    256
#define F_    263
#define FPAD  288
#define C_    512
#define D_    256
#define H_    4
#define HD_   64
#define LYR_  4

typedef __attribute__((ext_vector_type(16))) _Float16 v16h;
typedef __attribute__((ext_vector_type(8)))  float    v8f;

union Frag { v16h h; uint4 q[2]; };

__device__ __forceinline__ v8f wmma_f16(v16h a, v16h b, v8f c) {
  return __builtin_amdgcn_wmma_f32_16x16x32_f16(false, a, false, b, (short)0, c, false, false);
}

// ---------------------------------------------------------------------------
// Weight pre-pack into WMMA B-fragment layout.
// dst dword index = ((kt*(N/16)+nt)*32 + lane)*8 + v
// lane<16: col n = nt*16+lane, K pair = kt*32 + 2v, +1
// lane>=16: col n = nt*16+lane-16, K pair = kt*32 + 16 + 2v, +1
// ---------------------------------------------------------------------------
__global__ void pack_w(const float* __restrict__ W, uint32_t* __restrict__ dst,
                       int Ksrc, int Kpad, int N) {
  int tid = blockIdx.x * 256 + threadIdx.x;
  int total = (Kpad >> 5) * (N >> 4) * 256;
  if (tid >= total) return;
  int v    = tid & 7;
  int lane = (tid >> 3) & 31;
  int rest = tid >> 8;
  int nt16 = N >> 4;
  int nt   = rest % nt16;
  int kt   = rest / nt16;
  int n    = nt * 16 + (lane & 15);
  int k0   = kt * 32 + (lane >> 4) * 16 + 2 * v;
  float f0 = (k0     < Ksrc) ? W[(size_t)k0 * N + n]       : 0.f;
  float f1 = (k0 + 1 < Ksrc) ? W[(size_t)(k0 + 1) * N + n] : 0.f;
  _Float16 h0 = (_Float16)f0, h1 = (_Float16)f1;
  uint32_t lo = *(const uint16_t*)&h0;
  uint32_t hi = *(const uint16_t*)&h1;
  dst[tid] = lo | (hi << 16);
}

// f32 -> f16 with optional column zero-padding
__global__ void cvt_pad_f16(const float* __restrict__ src, _Float16* __restrict__ dst,
                            int rows, int csrc, int cdst) {
  long long tid = (long long)blockIdx.x * 256 + threadIdx.x;
  long long total = (long long)rows * cdst;
  if (tid >= total) return;
  int r = (int)(tid / cdst), c = (int)(tid % cdst);
  dst[tid] = (c < csrc) ? (_Float16)src[(size_t)r * csrc + c] : (_Float16)0.f;
}

// ---------------------------------------------------------------------------
// Generic WMMA GEMM: out = epilogue(A[M,K] @ W[K,N] + bias)
// block = 256 threads (8 waves); wave -> 16 rows x 128 cols (8 accumulators)
// block -> 128 rows x 128 cols
// ---------------------------------------------------------------------------
enum { MODE_F16 = 0, MODE_SILU = 1, MODE_GELU = 2, MODE_F32 = 3, MODE_RES = 4, MODE_FRAME = 5 };

template <int MODE>
__global__ void gemm_wmma(const _Float16* __restrict__ A, const uint4* __restrict__ Wp,
                          const float* __restrict__ bias,
                          _Float16* __restrict__ out16, float* __restrict__ out32,
                          const float* __restrict__ pos_emb, const float* __restrict__ cond_emb,
                          int M, int K, int N) {
  const int lane = threadIdx.x & 31;
  const int wave = threadIdx.x >> 5;
  const int mt   = blockIdx.x * 8 + wave;
  const int nb   = blockIdx.y;
  const int row0 = mt * 16;
  if (row0 >= M) return;
  const int l15 = lane & 15;
  const int lhi = lane >> 4;
  const int arow = row0 + l15;
  const int acol = lhi * 16;
  const int ntiles = N >> 4;

  v8f acc[8];
#pragma unroll
  for (int i = 0; i < 8; ++i)
#pragma unroll
    for (int p = 0; p < 8; ++p) acc[i][p] = 0.f;

  const int ksteps = K >> 5;
  for (int kt = 0; kt < ksteps; ++kt) {
    Frag a;
    const _Float16* ap = A + (size_t)arow * K + kt * 32 + acol;
    a.q[0] = *(const uint4*)(ap);
    a.q[1] = *(const uint4*)(ap + 8);
    __builtin_prefetch(ap + 32, 0, 1);  // global_prefetch next K-slab
#pragma unroll
    for (int nt = 0; nt < 8; ++nt) {
      int ntile = nb * 8 + nt;
      const uint4* bp = Wp + ((size_t)(kt * ntiles + ntile) * 32 + lane) * 2;
      Frag b;
      b.q[0] = bp[0];
      b.q[1] = bp[1];
      acc[nt] = wmma_f16(a.h, b.h, acc[nt]);
    }
  }

#pragma unroll
  for (int nt = 0; nt < 8; ++nt) {
    int ncol = nb * 128 + nt * 16 + l15;
    float bv = bias[ncol];
#pragma unroll
    for (int j = 0; j < 8; ++j) {
      int m = lhi ? (8 + j) : j;
      size_t idx = (size_t)(row0 + m) * N + ncol;
      float v = acc[nt][j] + bv;
      if (MODE == MODE_F16) {
        out16[idx] = (_Float16)v;
      } else if (MODE == MODE_SILU) {
        out16[idx] = (_Float16)(v / (1.f + expf(-v)));
      } else if (MODE == MODE_GELU) {
        out16[idx] = (_Float16)(0.5f * v * (1.f + erff(v * 0.70710678118654752f)));
      } else if (MODE == MODE_F32) {
        out32[idx] = v;
      } else if (MODE == MODE_RES) {
        out32[idx] += v;
      } else {  // MODE_FRAME: + pos_emb[t] + cond_emb[b]
        int r = row0 + m;
        int t = r & (T_ - 1);
        int bb = r >> 8;
        out32[idx] = v + pos_emb[t * D_ + ncol] + cond_emb[(size_t)bb * D_ + ncol];
      }
    }
  }
}

// ---------------------------------------------------------------------------
// LayerNorm(h row of 256) -> f16. 8 waves/block, one row per wave (wave32).
// ---------------------------------------------------------------------------
__global__ void ln_f16(const float* __restrict__ h, const float* __restrict__ g,
                       const float* __restrict__ bta, _Float16* __restrict__ y, int M) {
  int lane = threadIdx.x & 31, wave = threadIdx.x >> 5;
  int row = blockIdx.x * 8 + wave;
  if (row >= M) return;
  const float* hp = h + (size_t)row * D_;
  float4 x0 = *(const float4*)(hp + lane * 8);
  float4 x1 = *(const float4*)(hp + lane * 8 + 4);
  float x[8] = {x0.x, x0.y, x0.z, x0.w, x1.x, x1.y, x1.z, x1.w};
  float s = 0.f;
#pragma unroll
  for (int k = 0; k < 8; ++k) s += x[k];
#pragma unroll
  for (int m = 1; m < 32; m <<= 1) s += __shfl_xor(s, m, 32);
  float mean = s * (1.f / 256.f);
  float ss = 0.f;
#pragma unroll
  for (int k = 0; k < 8; ++k) { float d = x[k] - mean; ss += d * d; }
#pragma unroll
  for (int m = 1; m < 32; m <<= 1) ss += __shfl_xor(ss, m, 32);
  float rs = rsqrtf(ss * (1.f / 256.f) + 1e-5f);
  union { uint4 q; _Float16 hh[8]; } o;
#pragma unroll
  for (int k = 0; k < 8; ++k) {
    int c = lane * 8 + k;
    o.hh[k] = (_Float16)((x[k] - mean) * rs * g[c] + bta[c]);
  }
  *(uint4*)(y + (size_t)row * D_ + lane * 8) = o.q;
}

// ---------------------------------------------------------------------------
// Attention: one block per (b, h). 8 waves; each wave does 2 q-tiles of 16 rows.
// Three passes to keep register pressure low (no scratch spills):
//  pass 1: S tiles via WMMA, one live accumulator, raw scaled scores -> LDS f16
//  pass 2: row softmax (bias applied here) reading/writing the LDS P tile
//  pass 3: O = P V via WMMA from LDS A/B fragments
// ---------------------------------------------------------------------------
#define KP 72     // K LDS pitch (halves)
#define VP 264    // Vt / P LDS pitch (halves)
#define SM_K  (256 * KP * 2)           // 36864
#define SM_V  (64 * VP * 2)            // 33792
#define SM_P  (8 * 16 * VP * 2)        // 67584
#define SM_BIAS (256 * 4)              // 1024
#define SM_TOTAL (SM_K + SM_V + SM_P + SM_BIAS)

__global__ void attn_wmma(const _Float16* __restrict__ qkv,
                          const unsigned char* __restrict__ mask,
                          _Float16* __restrict__ o16) {
  extern __shared__ __align__(16) char smem[];
  _Float16* Kls = (_Float16*)smem;
  _Float16* Vt  = (_Float16*)(smem + SM_K);
  _Float16* Pls = (_Float16*)(smem + SM_K + SM_V);
  float* biasL  = (float*)(smem + SM_K + SM_V + SM_P);

  const int bh = blockIdx.x;
  const int b  = bh >> 2;
  const int h  = bh & 3;
  const int tid = threadIdx.x;
  const int lane = tid & 31, wave = tid >> 5;
  const size_t rowbase = (size_t)b * T_;

  // stage K (row-major) and V (transposed) into LDS, 16B chunks
  for (int c = tid; c < 256 * 8; c += 256) {
    int t = c >> 3, ch = c & 7;
    uint4 kv = *(const uint4*)(qkv + (rowbase + t) * 768 + 256 + h * 64 + ch * 8);
    *(uint4*)&Kls[t * KP + ch * 8] = kv;
    union { uint4 q; _Float16 hh[8]; } vv;
    vv.q = *(const uint4*)(qkv + (rowbase + t) * 768 + 512 + h * 64 + ch * 8);
#pragma unroll
    for (int e = 0; e < 8; ++e) Vt[(ch * 8 + e) * VP + t] = vv.hh[e];
  }
  biasL[tid] = mask[rowbase + tid] ? 0.f : -1e9f;
  __syncthreads();

  const int l15 = lane & 15, lhi = lane >> 4;
  _Float16* Pw = Pls + wave * 16 * VP;

  for (int qi = 0; qi < 2; ++qi) {
    const int qt = wave + qi * 8;
    // Q A-fragments (16x64 -> 2 frags of 16x32)
    Frag qa[2];
#pragma unroll
    for (int kt = 0; kt < 2; ++kt) {
      const _Float16* qp = qkv + (rowbase + qt * 16 + l15) * 768 + h * 64 + kt * 32 + lhi * 16;
      qa[kt].q[0] = *(const uint4*)qp;
      qa[kt].q[1] = *(const uint4*)(qp + 8);
    }

    // ---- pass 1: S tiles, one accumulator live, stash scaled raw scores ----
#pragma unroll
    for (int nt = 0; nt < 16; ++nt) {
      v8f s;
#pragma unroll
      for (int p = 0; p < 8; ++p) s[p] = 0.f;
#pragma unroll
      for (int kt = 0; kt < 2; ++kt) {
        Frag bf;
        const _Float16* kp = &Kls[(nt * 16 + l15) * KP + kt * 32 + lhi * 16];
        bf.q[0] = *(const uint4*)kp;
        bf.q[1] = *(const uint4*)(kp + 8);
        s = wmma_f16(qa[kt].h, bf.h, s);
      }
#pragma unroll
      for (int j = 0; j < 8; ++j) {
        int mrow = lhi ? (8 + j) : j;
        Pw[mrow * VP + nt * 16 + l15] = (_Float16)(s[j] * 0.125f);
      }
    }

    // ---- pass 2: row softmax (same-wave LDS ops are in-order) ----
#pragma unroll
    for (int j = 0; j < 8; ++j) {
      int mrow = lhi ? (8 + j) : j;
      float vals[16];
      float mx = -3.0e38f;
#pragma unroll
      for (int nt = 0; nt < 16; ++nt) {
        float sv = (float)Pw[mrow * VP + nt * 16 + l15] + biasL[nt * 16 + l15];
        vals[nt] = sv;
        mx = fmaxf(mx, sv);
      }
#pragma unroll
      for (int m = 1; m < 16; m <<= 1) mx = fmaxf(mx, __shfl_xor(mx, m, 32));
      float sum = 0.f;
#pragma unroll
      for (int nt = 0; nt < 16; ++nt) { float e = expf(vals[nt] - mx); vals[nt] = e; sum += e; }
#pragma unroll
      for (int m = 1; m < 16; m <<= 1) sum += __shfl_xor(sum, m, 32);
      float inv = 1.f / sum;
#pragma unroll
      for (int nt = 0; nt < 16; ++nt) Pw[mrow * VP + nt * 16 + l15] = (_Float16)(vals[nt] * inv);
    }

    // ---- pass 3: O = P V ----
#pragma unroll
    for (int nt2 = 0; nt2 < 4; ++nt2) {
      v8f oacc;
#pragma unroll
      for (int p = 0; p < 8; ++p) oacc[p] = 0.f;
#pragma unroll
      for (int kc = 0; kc < 8; ++kc) {
        Frag pa, vb;
        const _Float16* pp = &Pw[l15 * VP + kc * 32 + lhi * 16];
        pa.q[0] = *(const uint4*)pp;
        pa.q[1] = *(const uint4*)(pp + 8);
        const _Float16* vp = &Vt[(nt2 * 16 + l15) * VP + kc * 32 + lhi * 16];
        vb.q[0] = *(const uint4*)vp;
        vb.q[1] = *(const uint4*)(vp + 8);
        oacc = wmma_f16(pa.h, vb.h, oacc);
      }
#pragma unroll
      for (int j = 0; j < 8; ++j) {
        int mrow = lhi ? (8 + j) : j;
        o16[(rowbase + qt * 16 + mrow) * 256 + h * 64 + nt2 * 16 + l15] = (_Float16)oacc[j];
      }
    }
  }
}

// ---------------------------------------------------------------------------
// Head sigmoid + endpoint-constrained budget top-k (stable rank = argsort match)
// One block per batch row; 256 threads = 256 time steps.
// ---------------------------------------------------------------------------
__global__ void head_topk(const float* __restrict__ h, const unsigned char* __restrict__ mask,
                          const float* __restrict__ head_w, const float* __restrict__ head_b,
                          float* __restrict__ out) {
  __shared__ float sw[256];
  __shared__ float sc[256];
  __shared__ int   Lred[256];
  int b = blockIdx.x, t = threadIdx.x;
  sw[t] = head_w[t];
  int valid = mask[(size_t)b * T_ + t] ? 1 : 0;
  Lred[t] = valid;
  __syncthreads();
  for (int s = 128; s > 0; s >>= 1) {
    if (t < s) Lred[t] += Lred[t + s];
    __syncthreads();
  }
  int L = Lred[0];

  const float* hp = h + ((size_t)b * T_ + t) * D_;
  float dot = 0.f;
  for (int d = 0; d < D_; ++d) dot += hp[d] * sw[d];
  float vf = valid ? 1.f : 0.f;
  float prob = (1.f / (1.f + expf(-(dot + head_b[0])))) * vf;

  int Lc = (L > 1) ? L : 1;
  int ep = (t == 0 || t == Lc - 1) ? 1 : 0;
  float probs_e = fmaxf(prob, (float)ep) * vf;
  int is_inner = (t >= 1 && t <= L - 2 && valid) ? 1 : 0;
  sc[t] = is_inner ? probs_e : -INFINITY;
  __syncthreads();

  float my = sc[t];
  int cnt = 0;
  for (int j = 0; j < 256; ++j) {
    float o = sc[j];
    cnt += (o > my) || (o == my && j < t);
  }
  int budget = (int)rintf((float)L * 0.1f);   // round-to-nearest-even == jnp.round
  budget = budget < 2 ? 2 : (budget > L ? L : budget);
  int inner_budget = (L > 2) ? (budget - 2 > 0 ? budget - 2 : 0) : 0;
  int hard = ((is_inner && cnt < inner_budget) || (ep && valid)) ? 1 : 0;

  out[(size_t)b * T_ + t] = probs_e;
  out[(size_t)B_ * T_ + (size_t)b * T_ + t] = (float)hard * vf;
}

// ---------------------------------------------------------------------------
// Host side
// ---------------------------------------------------------------------------
static inline size_t align256(size_t x) { return (x + 255) & ~(size_t)255; }

extern "C" void kernel_launch(void* const* d_in, const int* in_sizes, int n_in,
                              void* d_out, int out_size, void* d_ws, size_t ws_size,
                              hipStream_t stream) {
  const float* motion  = (const float*)d_in[0];
  const unsigned char* vmask = (const unsigned char*)d_in[1];
  const float* cond    = (const float*)d_in[2];
  const float* frame_w = (const float*)d_in[3];
  const float* frame_b = (const float*)d_in[4];
  const float* pos_emb = (const float*)d_in[5];
  const float* cond_w1 = (const float*)d_in[6];
  const float* cond_b1 = (const float*)d_in[7];
  const float* cond_w2 = (const float*)d_in[8];
  const float* cond_b2 = (const float*)d_in[9];
  const float* ln1_g   = (const float*)d_in[10];
  const float* ln1_b   = (const float*)d_in[11];
  const float* qkv_w   = (const float*)d_in[12];
  const float* qkv_b   = (const float*)d_in[13];
  const float* out_w   = (const float*)d_in[14];
  const float* out_b   = (const float*)d_in[15];
  const float* ln2_g   = (const float*)d_in[16];
  const float* ln2_b   = (const float*)d_in[17];
  const float* ff_w1   = (const float*)d_in[18];
  const float* ff_b1   = (const float*)d_in[19];
  const float* ff_w2   = (const float*)d_in[20];
  const float* ff_b2   = (const float*)d_in[21];
  const float* head_w  = (const float*)d_in[22];
  const float* head_b  = (const float*)d_in[23];

  const size_t BT = (size_t)B_ * T_;
  char* ws = (char*)d_ws;
  size_t off = 0;
  auto grab = [&](size_t bytes) { char* p = ws + off; off = align256(off + bytes); return p; };

  float*    hbuf   = (float*)   grab(BT * D_ * 4);          // residual stream f32
  _Float16* ybuf   = (_Float16*)grab(BT * D_ * 2);          // LN out / attn out f16
  _Float16* bigbuf = (_Float16*)grab(BT * 4 * D_ * 2);      // qkv (768) / ff hidden (1024)
  _Float16* mot16  = (_Float16*)grab(BT * FPAD * 2);        // padded motion f16
  _Float16* cond16 = (_Float16*)grab((size_t)B_ * C_ * 2);
  _Float16* cact   = (_Float16*)grab((size_t)B_ * D_ * 2);
  float*    cemb   = (float*)   grab((size_t)B_ * D_ * 4);

  uint32_t* framePk = (uint32_t*)grab((size_t)FPAD * D_ * 2);
  uint32_t* c1Pk    = (uint32_t*)grab((size_t)C_ * D_ * 2);
  uint32_t* c2Pk    = (uint32_t*)grab((size_t)D_ * D_ * 2);
  uint32_t* qkvPk   = (uint32_t*)grab((size_t)LYR_ * D_ * 3 * D_ * 2);
  uint32_t* outPk   = (uint32_t*)grab((size_t)LYR_ * D_ * D_ * 2);
  uint32_t* ff1Pk   = (uint32_t*)grab((size_t)LYR_ * D_ * 4 * D_ * 2);
  uint32_t* ff2Pk   = (uint32_t*)grab((size_t)LYR_ * 4 * D_ * D_ * 2);

  dim3 blk(256);

  // ---- prep: conversions & weight packing ----
  {
    long long tot = (long long)BT * FPAD;
    cvt_pad_f16<<<dim3((unsigned)((tot + 255) / 256)), blk, 0, stream>>>(motion, mot16, (int)BT, F_, FPAD);
    tot = (long long)B_ * C_;
    cvt_pad_f16<<<dim3((unsigned)((tot + 255) / 256)), blk, 0, stream>>>(cond, cond16, B_, C_, C_);
  }
  auto pack = [&](const float* W, uint32_t* dst, int Ksrc, int Kpad, int N) {
    int tot = (Kpad >> 5) * (N >> 4) * 256;
    pack_w<<<dim3((tot + 255) / 256), blk, 0, stream>>>(W, dst, Ksrc, Kpad, N);
  };
  pack(frame_w, framePk, F_, FPAD, D_);
  pack(cond_w1, c1Pk, C_, C_, D_);
  pack(cond_w2, c2Pk, D_, D_, D_);
  for (int i = 0; i < LYR_; ++i) {
    pack(qkv_w + (size_t)i * D_ * 3 * D_, qkvPk + (size_t)i * D_ * 3 * D_ / 2, D_, D_, 3 * D_);
    pack(out_w + (size_t)i * D_ * D_,     outPk + (size_t)i * D_ * D_ / 2,     D_, D_, D_);
    pack(ff_w1 + (size_t)i * D_ * 4 * D_, ff1Pk + (size_t)i * D_ * 4 * D_ / 2, D_, D_, 4 * D_);
    pack(ff_w2 + (size_t)i * 4 * D_ * D_, ff2Pk + (size_t)i * 4 * D_ * D_ / 2, 4 * D_, 4 * D_, D_);
  }

  // ---- cond MLP: silu(cond@w1+b1)@w2+b2 ----
  gemm_wmma<MODE_SILU><<<dim3(B_ / 128, D_ / 128), blk, 0, stream>>>(
      cond16, (const uint4*)c1Pk, cond_b1, cact, nullptr, nullptr, nullptr, B_, C_, D_);
  gemm_wmma<MODE_F32><<<dim3(B_ / 128, D_ / 128), blk, 0, stream>>>(
      cact, (const uint4*)c2Pk, cond_b2, nullptr, cemb, nullptr, nullptr, B_, D_, D_);

  // ---- frame embed: h = motion@frame_w + frame_b + pos + cond_emb ----
  gemm_wmma<MODE_FRAME><<<dim3((unsigned)(BT / 128), D_ / 128), blk, 0, stream>>>(
      mot16, (const uint4*)framePk, frame_b, nullptr, hbuf, pos_emb, cemb, (int)BT, FPAD, D_);

  // ---- transformer layers ----
  for (int i = 0; i < LYR_; ++i) {
    ln_f16<<<dim3((unsigned)(BT / 8)), blk, 0, stream>>>(hbuf, ln1_g + i * D_, ln1_b + i * D_, ybuf, (int)BT);
    gemm_wmma<MODE_F16><<<dim3((unsigned)(BT / 128), (3 * D_) / 128), blk, 0, stream>>>(
        ybuf, (const uint4*)(qkvPk + (size_t)i * D_ * 3 * D_ / 2), qkv_b + i * 3 * D_,
        bigbuf, nullptr, nullptr, nullptr, (int)BT, D_, 3 * D_);
    attn_wmma<<<dim3(B_ * H_), blk, SM_TOTAL, stream>>>(bigbuf, vmask, ybuf);
    gemm_wmma<MODE_RES><<<dim3((unsigned)(BT / 128), D_ / 128), blk, 0, stream>>>(
        ybuf, (const uint4*)(outPk + (size_t)i * D_ * D_ / 2), out_b + i * D_,
        nullptr, hbuf, nullptr, nullptr, (int)BT, D_, D_);
    ln_f16<<<dim3((unsigned)(BT / 8)), blk, 0, stream>>>(hbuf, ln2_g + i * D_, ln2_b + i * D_, ybuf, (int)BT);
    gemm_wmma<MODE_GELU><<<dim3((unsigned)(BT / 128), (4 * D_) / 128), blk, 0, stream>>>(
        ybuf, (const uint4*)(ff1Pk + (size_t)i * D_ * 4 * D_ / 2), ff_b1 + i * 4 * D_,
        bigbuf, nullptr, nullptr, nullptr, (int)BT, D_, 4 * D_);
    gemm_wmma<MODE_RES><<<dim3((unsigned)(BT / 128), D_ / 128), blk, 0, stream>>>(
        bigbuf, (const uint4*)(ff2Pk + (size_t)i * 4 * D_ * D_ / 2), ff_b2 + i * D_,
        nullptr, hbuf, nullptr, nullptr, (int)BT, 4 * D_, D_);
  }

  // ---- head + straight-through top-k ----
  head_topk<<<dim3(B_), blk, 0, stream>>>(hbuf, vmask, head_w, head_b, (float*)d_out);

  (void)in_sizes; (void)n_in; (void)out_size; (void)ws_size;
}
